// Attention_3478923510454
// MI455X (gfx1250) — matrix operand order repaired and checked by reading, not compiled, not run on hardware
//
#include <hip/hip_runtime.h>
#include <hip/hip_bf16.h>

// ---------------------------------------------------------------------------
// CDNA5 (gfx1250) implementation of the multi-head knowledge attention
// reference. Heavy GEMMs run on v_wmma_f32_16x16x32_bf16; LDS tile staging
// uses the Tensor Data Mover (TENSOR_LOAD_TO_LDS + s_wait_tensorcnt) with a
// double-buffered pipeline.
// ---------------------------------------------------------------------------

typedef __attribute__((ext_vector_type(16))) __bf16        v16bf;
typedef __attribute__((ext_vector_type(8)))  float         v8f;
typedef __attribute__((ext_vector_type(4)))  unsigned int  u32x4;
typedef __attribute__((ext_vector_type(8)))  int           i32x8;
typedef __attribute__((ext_vector_type(4)))  int           i32x4;

#if defined(__has_builtin)
#  if __has_builtin(__builtin_amdgcn_tensor_load_to_lds)
#    define ATTN_HAVE_TDM 1
#  endif
#endif
#ifndef ATTN_HAVE_TDM
#  define ATTN_HAVE_TDM 0
#endif

__device__ __forceinline__ unsigned short f2bf(float f) {
  unsigned int u = __float_as_uint(f);
  u += 0x7fffu + ((u >> 16) & 1u);      // round-to-nearest-even
  return (unsigned short)(u >> 16);
}

union FragAB   { v16bf v; unsigned int u[8]; };
union Vec16U16 { uint4 q[2]; unsigned short s[16]; };

// Low 32 bits of a generic pointer into LDS = wave-relative LDS byte address.
__device__ __forceinline__ unsigned lds_off_of(const void* p) {
  return (unsigned)(unsigned long long)p;
}

#if ATTN_HAVE_TDM
// Issue one TDM descriptor: load a 2D tile of 16-bit elements
// (tileX contiguous elems per row, tileY rows, row stride strideElems)
// from global memory into LDS, packed contiguously (row-major [tileY][tileX]).
// D# field packing per CDNA5 ISA "08_async_tensor" §8.3 / §8.4.
// This toolchain's builtin takes 6 operands:
//   (g0: uint32x4, g1: int32x8, g2: int32x4, g3: int32x4, g?: int32x8, cpol)
// groups 2/3 (tensor dims 2..4 / gather indices) are unused for 2D tiles.
__device__ __forceinline__ void tdm_load_tile_b16(unsigned ldsOff, const void* g,
                                                  unsigned tileX, unsigned tileY,
                                                  unsigned long long strideElems)
{
  const unsigned long long ga = (unsigned long long)g;
  u32x4 g0 = { 1u,                                             // count=1 (valid)
               ldsOff,                                         // lds_addr
               (unsigned)(ga & 0xffffffffu),                   // global_addr[31:0]
               ((unsigned)((ga >> 32) & 0x01ffffffu)) | 0x80000000u }; // addr[56:32] | type=2
  i32x8 g1;
  g1[0] = (int)(1u << 16);                                     // data_size = 2 bytes
  g1[1] = (int)((tileX & 0xffffu) << 16);                      // tensor_dim0[15:0]
  g1[2] = (int)(((tileX >> 16) & 0xffffu) |                    // tensor_dim0[31:16]
                ((tileY & 0xffffu) << 16));                    // tensor_dim1[15:0]
  g1[3] = (int)(((tileY >> 16) & 0xffffu) |                    // tensor_dim1[31:16]
                ((tileX & 0xffffu) << 16));                    // tile_dim0
  g1[4] = (int)(tileY & 0xffffu);                              // tile_dim1 (tile_dim2=0)
  g1[5] = (int)(unsigned)(strideElems & 0xffffffffu);          // tensor_dim0_stride[31:0]
  g1[6] = (int)(unsigned)((strideElems >> 32) & 0xffffu);      // stride[47:32]
  g1[7] = 0;
  i32x4 gz4 = { 0, 0, 0, 0 };
  i32x8 gz8 = { 0, 0, 0, 0, 0, 0, 0, 0 };
  __builtin_amdgcn_tensor_load_to_lds(g0, g1, gz4, gz4, gz8, 0);
}
#endif

// ---------------------------------------------------------------------------
// Stage one 128x32 A tile (as [m][k]) and one 32x128 B tile (as [n][k]) into
// LDS. bf16 A tiles and N-major ("transposed") B tiles go through the TDM;
// f32 A (attn re-read, needs conversion) and K-major B (needs transpose
// scatter) go through the per-lane VGPR path.
// ---------------------------------------------------------------------------
template<bool A_F32, bool B_TRANS>
__device__ __forceinline__ void stage_tiles(unsigned short* At, unsigned short* Bt,
    const unsigned short* __restrict__ Abf, const float* __restrict__ Af32,
    const unsigned short* __restrict__ Bp,
    long lda, long ldb, long blockM, long blockN, int kk, int tid)
{
  constexpr bool TDMA = (ATTN_HAVE_TDM != 0) && !A_F32;
  constexpr bool TDMB = (ATTN_HAVE_TDM != 0) && B_TRANS;
#if ATTN_HAVE_TDM
  if ((TDMA || TDMB) && tid < 32) {          // one wave issues the DMA
    if (TDMA)
      tdm_load_tile_b16(lds_off_of(At), Abf + blockM * lda + kk,
                        32u, 128u, (unsigned long long)lda);
    if (TDMB)
      tdm_load_tile_b16(lds_off_of(Bt), Bp + blockN * ldb + kk,
                        32u, 128u, (unsigned long long)ldb);
  }
#endif
  if (A_F32) {
    const int row = tid >> 1;                // 0..127
    const int seg = tid & 1;                 // 16 elems each
    const float* src = Af32 + (blockM + row) * lda + kk + seg * 16;
    unsigned int p[8];
#pragma unroll
    for (int i = 0; i < 8; ++i)
      p[i] = (unsigned int)f2bf(src[2 * i]) |
             ((unsigned int)f2bf(src[2 * i + 1]) << 16);
    uint4* d4 = (uint4*)&At[row * 32 + seg * 16];
    d4[0] = make_uint4(p[0], p[1], p[2], p[3]);
    d4[1] = make_uint4(p[4], p[5], p[6], p[7]);
  } else if (!TDMA) {
    const int row = tid >> 1;
    const int seg = tid & 1;
    const uint4* s4 = (const uint4*)(Abf + (blockM + row) * lda + kk + seg * 16);
    uint4* d4 = (uint4*)&At[row * 32 + seg * 16];
    d4[0] = s4[0];
    d4[1] = s4[1];
  }
  if (!B_TRANS) {
    // source [K x N] row-major -> transpose into LDS [n][k]
    const int k   = tid >> 3;                // 0..31
    const int seg = tid & 7;                 // 16 n's each
    Vec16U16 t;
    const uint4* s4 = (const uint4*)(Bp + (long)(kk + k) * ldb + blockN + seg * 16);
    t.q[0] = s4[0];
    t.q[1] = s4[1];
#pragma unroll
    for (int i = 0; i < 16; ++i)
      Bt[(seg * 16 + i) * 32 + k] = t.s[i];
  } else if (!TDMB) {
    const int n   = tid >> 1;
    const int seg = tid & 1;
    const uint4* s4 = (const uint4*)(Bp + (blockN + n) * ldb + kk + seg * 16);
    uint4* d4 = (uint4*)&Bt[n * 32 + seg * 16];
    d4[0] = s4[0];
    d4[1] = s4[1];
  }
}

// ---------------------------------------------------------------------------
// Generic batched GEMM: C[z] = act( scale * A[z] * B[z] + bias[z] )
//   A: [M x K] bf16 row-major (or f32 row-major when A_F32)
//   B: [K x N] bf16 row-major, or when B_TRANS the source is [N x K] row-major
// Block tile 128x128, 256 threads (8 waves, 2x4), K step 32, double-buffered
// LDS with TDM prefetch of the next K-slab overlapping the WMMAs.
// All problem dims are multiples of the tile sizes, so no bounds checks.
// ---------------------------------------------------------------------------
template<bool A_F32, bool B_TRANS, bool RELU, bool OUT_BF16>
__global__ __launch_bounds__(256)
void gemm_wmma_bf16(const void* __restrict__ Av, long lda, long strideA,
                    const unsigned short* __restrict__ Bv, long ldb, long strideB,
                    void* __restrict__ Cv, long ldc, long strideC,
                    const float* __restrict__ bias, long strideBias,
                    int Kdim, float scale)
{
  __shared__ __align__(16) unsigned short Atile[2][128 * 32];  // [m][k]
  __shared__ __align__(16) unsigned short Btile[2][128 * 32];  // [n][k]

  constexpr bool USE_TDM = (ATTN_HAVE_TDM != 0) && (!A_F32 || B_TRANS);

  const int  tid  = threadIdx.x;
  const int  lane = tid & 31;
  const int  wave = tid >> 5;
  const int  wm   = wave >> 2;                 // 0..1 (64 rows each)
  const int  wn   = wave & 3;                  // 0..3 (32 cols each)
  const long z    = blockIdx.z;
  const long blockM = (long)blockIdx.x * 128;
  const long blockN = (long)blockIdx.y * 128;

  const unsigned short* Abf   = (const unsigned short*)Av + (A_F32 ? 0 : z * strideA);
  const float*          Af32  = (const float*)Av + (A_F32 ? z * strideA : 0);
  const unsigned short* Bp    = Bv + z * strideB;
  const float*          biasp = bias ? (bias + z * strideBias) : nullptr;

  v8f acc[4][2];
#pragma unroll
  for (int mt = 0; mt < 4; ++mt)
#pragma unroll
    for (int nt = 0; nt < 2; ++nt)
      acc[mt][nt] = (v8f){0.f, 0.f, 0.f, 0.f, 0.f, 0.f, 0.f, 0.f};

  const int mlane = lane & 15;        // row (A) / col (B,C)
  const int khalf = lane >> 4;        // lane-half select

  // Prologue: stage K-slab 0 into buffer 0.
  stage_tiles<A_F32, B_TRANS>(Atile[0], Btile[0], Abf, Af32, Bp,
                              lda, ldb, blockM, blockN, 0, tid);
  if (USE_TDM) {
    if (tid < 32) __builtin_amdgcn_s_wait_tensorcnt(0);
  }
  __syncthreads();

  const int nsteps = Kdim >> 5;
  for (int s = 0; s < nsteps; ++s) {
    const int cur = s & 1;
    // Prefetch next K-slab into the other buffer while we compute.
    if (s + 1 < nsteps)
      stage_tiles<A_F32, B_TRANS>(Atile[cur ^ 1], Btile[cur ^ 1], Abf, Af32, Bp,
                                  lda, ldb, blockM, blockN, (s + 1) << 5, tid);

    // ---- load fragments from current buffer (contiguous k = 16*khalf + s, same for A and B) ----
    FragAB a[4], b[2];
#pragma unroll
    for (int mt = 0; mt < 4; ++mt) {
      const unsigned short* ar = &Atile[cur][(wm * 64 + mt * 16 + mlane) * 32];
#pragma unroll
      for (int j = 0; j < 8; ++j) {
        const int kp = khalf * 16 + 2 * j;
        a[mt].u[j] = *(const unsigned int*)&ar[kp];
      }
    }
#pragma unroll
    for (int nt = 0; nt < 2; ++nt) {
      const unsigned short* br =
          &Btile[cur][(wn * 32 + nt * 16 + mlane) * 32 + khalf * 16];
#pragma unroll
      for (int j = 0; j < 8; ++j)
        b[nt].u[j] = *(const unsigned int*)&br[2 * j];
    }

    // ---- 8 WMMAs per wave per K-step ----
#pragma unroll
    for (int mt = 0; mt < 4; ++mt)
#pragma unroll
      for (int nt = 0; nt < 2; ++nt)
        acc[mt][nt] = __builtin_amdgcn_wmma_f32_16x16x32_bf16(
            false, a[mt].v, false, b[nt].v, (short)0, acc[mt][nt], false, false);

    // Staging for s+1 must be complete before anyone reads it next iteration.
    if (USE_TDM) {
      if (tid < 32) __builtin_amdgcn_s_wait_tensorcnt(0);
    }
    __syncthreads();
  }

  // ---- epilogue ----
  const int mrow0 = khalf * 8;
#pragma unroll
  for (int mt = 0; mt < 4; ++mt) {
#pragma unroll
    for (int nt = 0; nt < 2; ++nt) {
      const long col = blockN + wn * 32 + nt * 16 + mlane;
      const float bv = biasp ? biasp[col] : 0.f;
#pragma unroll
      for (int j = 0; j < 8; ++j) {
        const long row = blockM + wm * 64 + mt * 16 + mrow0 + j;
        float v = acc[mt][nt][j] * scale + bv;
        if (RELU) v = fmaxf(v, 0.f);
        const long idx = z * strideC + row * ldc + col;
        if (OUT_BF16) ((unsigned short*)Cv)[idx] = f2bf(v);
        else          ((float*)Cv)[idx] = v;
      }
    }
  }
}

// ---------------------------------------------------------------------------
// Row softmax over 4096 entries, in place, with post-softmax multiplicative
// knowledge mask (reference semantics: mask applied AFTER softmax, no renorm).
// One 256-thread block per row; 16 elements per thread.
// ---------------------------------------------------------------------------
__global__ __launch_bounds__(256)
void softmax_mask_kernel(float* __restrict__ w, const float* __restrict__ mask)
{
  __shared__ float red[256];
  const int tid = threadIdx.x;
  float* p = w + (size_t)blockIdx.x * 4096;

  float vals[16];
  float mx = -3.4e38f;
#pragma unroll
  for (int j = 0; j < 16; ++j) {
    vals[j] = p[tid + j * 256];
    mx = fmaxf(mx, vals[j]);
  }
  red[tid] = mx;
  __syncthreads();
  for (int s = 128; s > 0; s >>= 1) {
    if (tid < s) red[tid] = fmaxf(red[tid], red[tid + s]);
    __syncthreads();
  }
  const float rowmax = red[0];
  __syncthreads();

  float sum = 0.f;
#pragma unroll
  for (int j = 0; j < 16; ++j) {
    vals[j] = __expf(vals[j] - rowmax);
    sum += vals[j];
  }
  red[tid] = sum;
  __syncthreads();
  for (int s = 128; s > 0; s >>= 1) {
    if (tid < s) red[tid] += red[tid + s];
    __syncthreads();
  }
  const float inv = 1.f / red[0];

#pragma unroll
  for (int j = 0; j < 16; ++j)
    p[tid + j * 256] = vals[j] * inv * mask[tid + j * 256];
}

// ---------------------------------------------------------------------------
__global__ __launch_bounds__(256)
void f32_to_bf16_kernel(const float* __restrict__ in,
                        unsigned short* __restrict__ out, long n)
{
  long i = (long)blockIdx.x * blockDim.x + threadIdx.x;
  const long stride = (long)gridDim.x * blockDim.x;
  for (; i < n; i += stride) out[i] = f2bf(in[i]);
}

// ---------------------------------------------------------------------------
extern "C" void kernel_launch(void* const* d_in, const int* in_sizes, int n_in,
                              void* d_out, int out_size, void* d_ws, size_t ws_size,
                              hipStream_t stream)
{
  (void)in_sizes; (void)n_in; (void)out_size; (void)ws_size;

  const float* sentences = (const float*)d_in[0];   // [4096,1024]
  const float* knowledge = (const float*)d_in[1];   // [4096,1024]
  const float* mask      = (const float*)d_in[2];   // [4096]
  const float* Ws        = (const float*)d_in[3];   // [8,1024,128]
  const float* bs        = (const float*)d_in[4];   // [8,128]
  const float* Wk        = (const float*)d_in[5];   // [8,1024,128]
  const float* bk        = (const float*)d_in[6];   // [8,128]
  const float* W1        = (const float*)d_in[7];   // [8192,1024]
  const float* b1        = (const float*)d_in[8];   // [1024]
  const float* W2        = (const float*)d_in[9];   // [1024,1024]
  const float* b2        = (const float*)d_in[10];  // [1024]

  float* out     = (float*)d_out;                   // [4096,1024]
  float* weights = out + (size_t)4096 * 1024;       // [8*4096, 4096]

  // --- workspace carve (256B aligned) ---
  char* ws = (char*)d_ws;
  size_t off = 0;
  auto carve = [&](size_t bytes) -> char* {
    char* p = ws + off;
    off = (off + bytes + 255) & ~(size_t)255;
    return p;
  };
  unsigned short* sent_bf   = (unsigned short*)carve((size_t)4096 * 1024 * 2);
  unsigned short* know_bf   = (unsigned short*)carve((size_t)4096 * 1024 * 2);
  unsigned short* Ws_bf     = (unsigned short*)carve((size_t)8 * 1024 * 128 * 2);
  unsigned short* Wk_bf     = (unsigned short*)carve((size_t)8 * 1024 * 128 * 2);
  unsigned short* W1_bf     = (unsigned short*)carve((size_t)8192 * 1024 * 2);
  unsigned short* W2_bf     = (unsigned short*)carve((size_t)1024 * 1024 * 2);
  unsigned short* S_bf      = (unsigned short*)carve((size_t)8 * 4096 * 128 * 2);
  unsigned short* K_bf      = (unsigned short*)carve((size_t)8 * 4096 * 128 * 2);
  unsigned short* concat_bf = (unsigned short*)carve((size_t)4096 * 8192 * 2);
  unsigned short* h_bf      = (unsigned short*)carve((size_t)4096 * 1024 * 2);

  const dim3 blk(256);

  // --- f32 -> bf16 conversions ---
  f32_to_bf16_kernel<<<2048, blk, 0, stream>>>(sentences, sent_bf, 4096L * 1024);
  f32_to_bf16_kernel<<<2048, blk, 0, stream>>>(knowledge, know_bf, 4096L * 1024);
  f32_to_bf16_kernel<<<1024, blk, 0, stream>>>(Ws, Ws_bf, 8L * 1024 * 128);
  f32_to_bf16_kernel<<<1024, blk, 0, stream>>>(Wk, Wk_bf, 8L * 1024 * 128);
  f32_to_bf16_kernel<<<2048, blk, 0, stream>>>(W1, W1_bf, 8192L * 1024);
  f32_to_bf16_kernel<<<1024, blk, 0, stream>>>(W2, W2_bf, 1024L * 1024);

  // --- S[h] = sentences @ Ws[h] + bs[h]  (bf16 out, [8,4096,128]) ---
  gemm_wmma_bf16<false, false, false, true><<<dim3(32, 1, 8), blk, 0, stream>>>(
      sent_bf, 1024, 0, Ws_bf, 128, 1024L * 128,
      S_bf, 128, 4096L * 128, bs, 128, /*K=*/1024, 1.0f);

  // --- K[h] = knowledge @ Wk[h] + bk[h] ---
  gemm_wmma_bf16<false, false, false, true><<<dim3(32, 1, 8), blk, 0, stream>>>(
      know_bf, 1024, 0, Wk_bf, 128, 1024L * 128,
      K_bf, 128, 4096L * 128, bk, 128, /*K=*/1024, 1.0f);

  // --- scores[h] = (S[h] @ K[h]^T) / sqrt(128)  (f32, into weights output) ---
  gemm_wmma_bf16<false, true, false, false><<<dim3(32, 32, 8), blk, 0, stream>>>(
      S_bf, 128, 4096L * 128, K_bf, 128, 4096L * 128,
      weights, 4096, 4096L * 4096, nullptr, 0, /*K=*/128, 0.08838834764831845f);

  // --- softmax over knowledge dim, then post-softmax mask, in place ---
  softmax_mask_kernel<<<8 * 4096, blk, 0, stream>>>(weights, mask);

  // --- head_out[h] = attn[h] @ knowledge, written as concat[n, h*1024 + d] ---
  gemm_wmma_bf16<true, false, false, true><<<dim3(32, 8, 8), blk, 0, stream>>>(
      weights, 4096, 4096L * 4096, know_bf, 1024, 0,
      concat_bf, 8192, 1024, nullptr, 0, /*K=*/4096, 1.0f);

  // --- h = relu(concat @ W1 + b1) ---
  gemm_wmma_bf16<false, false, true, true><<<dim3(32, 8, 1), blk, 0, stream>>>(
      concat_bf, 8192, 0, W1_bf, 1024, 0,
      h_bf, 1024, 0, b1, 0, /*K=*/8192, 1.0f);

  // --- output = h @ W2 + b2  (f32 into d_out) ---
  gemm_wmma_bf16<false, false, false, false><<<dim3(32, 8, 1), blk, 0, stream>>>(
      h_bf, 1024, 0, W2_bf, 1024, 0,
      out, 1024, 0, b2, 0, /*K=*/1024, 1.0f);
}
